// EquiMultiHeadAttention_41094247088137
// MI455X (gfx1250) — matrix operand order repaired and checked by reading, not compiled, not run on hardware
//
#include <hip/hip_runtime.h>
#include <hip/hip_bf16.h>

typedef __attribute__((ext_vector_type(16))) _Float16 v16h;
typedef __attribute__((ext_vector_type(8)))  _Float16 v8h;
typedef __attribute__((ext_vector_type(8)))  float    v8f;

#define NB 4         // batch
#define SEQ 2048     // sequence
#define NH 8         // heads
#define HC 16        // hidden channels
#define MV 16        // multivector components
#define DFULL 256    // HC*MV
#define DC 128       // HC * 8 surviving components
// SCALE = 1/sqrt(32); fold log2(e) into Q so softmax uses exp2 (v_exp_f32)
#define QSCALE (0.17677669529663687f * 1.4426950408889634f)

// LDS row strides (elements): byte strides 272 / 80 are == 16 (mod 32) so the
// 16 lanes' 16B ds_load_b128 reads tile all 64 LDS banks without conflict.
#define KSTRIDE 136
#define VSTRIDE 40

__device__ __constant__ int c_xs[8] = {0, 2, 3, 4, 8, 9, 10, 14};

// ---------------------------------------------------------------------------
// Kernel 1: QKV projection. x:[B,S,16,16]f32 -> Qc/Kc:[B,H,S,128]f16 (masked,
// Q pre-scaled by SCALE*log2e), Vt:[B,H,256,S]f16 (dim-major for WMMA A-frags)
// ---------------------------------------------------------------------------
__global__ void k1_qkvproj(const float* __restrict__ x,
                           const float* __restrict__ Wq,
                           const float* __restrict__ bq,
                           _Float16* __restrict__ Qc,
                           _Float16* __restrict__ Kc,
                           _Float16* __restrict__ Vt) {
    const int which = threadIdx.x >> 7;          // token slot in block (0/1)
    const int lid   = threadIdx.x & 127;         // (h,c) pair
    const int tok   = blockIdx.x * 2 + which;
    const int b = tok / SEQ, s = tok % SEQ;

    __shared__ float xt[2][DFULL];
    ((float2*)xt[which])[lid] = ((const float2*)(x + (size_t)tok * DFULL))[lid];
    __syncthreads();

    const int h = lid >> 4, c = lid & 15;
    const float* xx = xt[which];
    const size_t bh = (size_t)b * NH + h;

    #pragma unroll
    for (int p = 0; p < 3; ++p) {
        const int o = h * 48 + c * 3 + p;
        float acc[MV];
        #pragma unroll
        for (int xi = 0; xi < MV; ++xi) acc[xi] = 0.f;
        #pragma unroll
        for (int cc = 0; cc < HC; ++cc) {
            const float w = Wq[o * HC + cc];
            #pragma unroll
            for (int xi = 0; xi < MV; ++xi) acc[xi] += w * xx[cc * MV + xi];
        }
        acc[0] += bq[o];                         // bias on scalar blade only

        if (p == 0) {
            _Float16* dst = Qc + (bh * SEQ + s) * DC + c * 8;
            #pragma unroll
            for (int mi = 0; mi < 8; ++mi)
                dst[mi] = (_Float16)(acc[c_xs[mi]] * QSCALE);
        } else if (p == 1) {
            _Float16* dst = Kc + (bh * SEQ + s) * DC + c * 8;
            #pragma unroll
            for (int mi = 0; mi < 8; ++mi)
                dst[mi] = (_Float16)acc[c_xs[mi]];
        } else {
            #pragma unroll
            for (int xi = 0; xi < MV; ++xi)
                Vt[(bh * DFULL + c * MV + xi) * SEQ + s] = (_Float16)acc[xi];
        }
    }
}

// ---------------------------------------------------------------------------
// Async prefetch of one 32-key stage (K: 32x128 f16, V^T: 256x32 f16) into
// padded LDS, cooperatively across the 128 threads of the block. Each
// global_load_async_to_lds_b128 moves 16B/lane, tracked by ASYNCcnt.
// ---------------------------------------------------------------------------
__device__ __forceinline__ void prefetch_stage(const _Float16* __restrict__ Kg,
                                               const _Float16* __restrict__ Vg,
                                               _Float16* lk, _Float16* lv,
                                               int kb, int tid) {
    // K tile: 512 chunks of 16B (32 rows x 256B), LDS row stride 272B
    #pragma unroll
    for (int it = 0; it < 4; ++it) {
        const int i = tid + it * 128;
        const int row = i >> 4, col = i & 15;
        const unsigned lo =
            (unsigned)(uintptr_t)(lk + (size_t)row * KSTRIDE) + col * 16;
        const unsigned long long ga =
            (unsigned long long)(uintptr_t)(Kg + (size_t)(kb + row) * DC) + col * 16;
        asm volatile("global_load_async_to_lds_b128 %0, %1, off"
                     :: "v"(lo), "v"(ga) : "memory");
    }
    // V tile: 1024 chunks of 16B (256 rows x 64B), LDS row stride 80B
    #pragma unroll
    for (int it = 0; it < 8; ++it) {
        const int i = tid + it * 128;
        const int row = i >> 2, col = i & 3;
        const unsigned lo =
            (unsigned)(uintptr_t)(lv + (size_t)row * VSTRIDE) + col * 16;
        const unsigned long long ga =
            (unsigned long long)(uintptr_t)(Vg + (size_t)row * SEQ + kb) + col * 16;
        asm volatile("global_load_async_to_lds_b128 %0, %1, off"
                     :: "v"(lo), "v"(ga) : "memory");
    }
}

// ---------------------------------------------------------------------------
// Kernel 2: flash attention, S^T = K*Q^T formulation, O^T = V^T*P^T.
// 1 wave = 16 queries, 4 waves/block sharing K/V stages via async-LDS
// double buffering. Full pass over 2048 keys with online softmax.
// ---------------------------------------------------------------------------
__global__ __launch_bounds__(128) void k2_attn(const _Float16* __restrict__ Qc,
                                               const _Float16* __restrict__ Kc,
                                               const _Float16* __restrict__ Vt,
                                               _Float16* __restrict__ Ot) {
    const int nqc   = SEQ / 64;
    const int bh    = blockIdx.x / nqc;
    const int qch   = blockIdx.x % nqc;
    const int lane  = threadIdx.x & 31;
    const bool hi   = lane >= 16;
    const int  ln   = lane & 15;
    const int jbase = qch * 64 + (threadIdx.x >> 5) * 16;
    const int tid   = threadIdx.x;

    __shared__ __align__(16) _Float16 ldsK[2][32 * KSTRIDE];   // 17.0 KB
    __shared__ __align__(16) _Float16 ldsV[2][DFULL * VSTRIDE]; // 40.0 KB

    // Q^T B-fragments: lane = query column, contiguous 16 f16 per d-chunk
    v16h qf[4];
    const _Float16* Qrow = Qc + ((size_t)bh * SEQ + jbase + ln) * DC;
    #pragma unroll
    for (int dk = 0; dk < 4; ++dk)
        qf[dk] = *(const v16h*)(Qrow + dk * 32 + (hi ? 16 : 0));

    const _Float16* Kbase = Kc + (size_t)bh * SEQ * DC;
    const _Float16* Vbase = Vt + (size_t)bh * DFULL * SEQ;

    v8f oacc[16];
    #pragma unroll
    for (int n = 0; n < 16; ++n)
        #pragma unroll
        for (int r = 0; r < 8; ++r) oacc[n][r] = 0.f;

    float m = -1e30f, l = 0.f;

    prefetch_stage(Kbase, Vbase, ldsK[0], ldsV[0], 0, tid);

    for (int g = 0; g < SEQ / 32; ++g) {
        const int buf = g & 1;
        asm volatile("s_wait_asynccnt 0x0" ::: "memory");
        __syncthreads();                          // stage `buf` ready for all
        if (g + 1 < SEQ / 32)
            prefetch_stage(Kbase, Vbase, ldsK[buf ^ 1], ldsV[buf ^ 1],
                           (g + 1) * 32, tid);

        const _Float16* lk = ldsK[buf];
        const _Float16* lv = ldsV[buf];

        // ---- S^T tiles for 32 keys: two 16x16 f32 accumulators ----
        v8f sacc[2];
        #pragma unroll
        for (int t = 0; t < 2; ++t) {
            v8f acc;
            #pragma unroll
            for (int r = 0; r < 8; ++r) acc[r] = 0.f;
            const _Float16* Krow = lk + (size_t)(t * 16 + ln) * KSTRIDE;
            #pragma unroll
            for (int dk = 0; dk < 4; ++dk) {
                v8h klo = *(const v8h*)(Krow + dk * 32 + (hi ? 16 : 0));
                v8h khi = *(const v8h*)(Krow + dk * 32 + 8 + (hi ? 16 : 0));
                v16h kf = __builtin_shufflevector(klo, khi,
                    0,1,2,3,4,5,6,7,8,9,10,11,12,13,14,15);
                acc = __builtin_amdgcn_wmma_f32_16x16x32_f16(
                    false, kf, false, qf[dk], (short)0, acc, false, false);
            }
            sacc[t] = acc;
        }

        // ---- online softmax: per-lane = one query column ----
        float mloc = -1e30f;
        #pragma unroll
        for (int r = 0; r < 8; ++r) {
            mloc = fmaxf(mloc, sacc[0][r]);
            mloc = fmaxf(mloc, sacc[1][r]);
        }
        mloc = fmaxf(mloc, __shfl_xor(mloc, 16));
        const float mnew = fmaxf(m, mloc);

        // wave-uniform skip of the O/l rescale when no lane's max moved
        if (__ballot(mnew > m) != 0ull) {
            const float corr = exp2f(m - mnew);
            l *= corr;
            #pragma unroll
            for (int n = 0; n < 16; ++n)
                #pragma unroll
                for (int r = 0; r < 8; ++r) oacc[n][r] *= corr;
            m = mnew;
        }

        float p1[8], p2[8], lsum = 0.f;
        #pragma unroll
        for (int r = 0; r < 8; ++r) {
            p1[r] = exp2f(sacc[0][r] - m);
            p2[r] = exp2f(sacc[1][r] - m);
            lsum += p1[r] + p2[r];
        }
        lsum += __shfl_xor(lsum, 16);
        l += lsum;

        // ---- rebuild P^T B-fragment (32 keys x 16 queries) via xor-16 ----
        v16h pf;
        #pragma unroll
        for (int r = 0; r < 8; ++r) {
            const float x1 = p1[r], x2 = p2[r];
            const float y1 = __shfl_xor(x1, 16);
            const float y2 = __shfl_xor(x2, 16);
            pf[r]     = (_Float16)(hi ? y2 : x1);   // k_local = r
            pf[r + 8] = (_Float16)(hi ? x2 : y1);   // k_local = r + 8
        }

        // ---- O^T += V^T * P^T over 16 dim-tiles ----
        #pragma unroll
        for (int n = 0; n < 16; ++n) {
            const _Float16* Vrow = lv + (size_t)(n * 16 + ln) * VSTRIDE;
            v8h vlo = *(const v8h*)(Vrow + (hi ? 16 : 0));
            v8h vhi = *(const v8h*)(Vrow + 8 + (hi ? 16 : 0));
            v16h vf = __builtin_shufflevector(vlo, vhi,
                0,1,2,3,4,5,6,7,8,9,10,11,12,13,14,15);
            oacc[n] = __builtin_amdgcn_wmma_f32_16x16x32_f16(
                false, vf, false, pf, (short)0, oacc[n], false, false);
        }
    }

    // ---- normalize and store O^T: [B,H,dim,S] f16 ----
    const float invl = 1.f / l;
    _Float16* Obase = Ot + (size_t)bh * DFULL * SEQ + jbase + ln;
    #pragma unroll
    for (int n = 0; n < 16; ++n)
        #pragma unroll
        for (int r = 0; r < 8; ++r) {
            const int d = n * 16 + r + (hi ? 8 : 0);
            Obase[(size_t)d * SEQ] = (_Float16)(oacc[n][r] * invl);
        }
}

// ---------------------------------------------------------------------------
// Kernel 3: output projection. out[b,s,o,x] = b_out[o]*(x==0)
//   + sum_{c,h} W_out[o, c*8+h] * Ot[b,h,c*16+x,s]
// ---------------------------------------------------------------------------
__global__ void k3_outproj(const _Float16* __restrict__ Ot,
                           const float* __restrict__ Wo,
                           const float* __restrict__ bo,
                           float* __restrict__ out) {
    const int t = blockIdx.x;
    const int b = t / SEQ, s = t % SEQ;
    __shared__ float ldso[128 * 16];   // [ch][x], ch = c*8 + h
    __shared__ float ldsw[16 * 128];   // W_out
    const int tid = threadIdx.x;

    for (int i = tid; i < 2048; i += 256) ldsw[i] = Wo[i];
    for (int i = tid; i < 2048; i += 256) {
        const int ch = i >> 4, xi = i & 15;
        const int c = ch >> 3, h = ch & 7;
        ldso[i] = (float)Ot[((size_t)(b * NH + h) * DFULL + c * MV + xi) * SEQ + s];
    }
    __syncthreads();

    const int o = tid >> 4, xi = tid & 15;
    float acc = (xi == 0) ? bo[o] : 0.f;
    #pragma unroll 8
    for (int ch = 0; ch < 128; ++ch)
        acc += ldsw[o * 128 + ch] * ldso[ch * 16 + xi];
    out[((size_t)t * HC + o) * MV + xi] = acc;
}

// ---------------------------------------------------------------------------
extern "C" void kernel_launch(void* const* d_in, const int* in_sizes, int n_in,
                              void* d_out, int out_size, void* d_ws, size_t ws_size,
                              hipStream_t stream) {
    const float* x  = (const float*)d_in[0];
    const float* Wq = (const float*)d_in[1];
    const float* bq = (const float*)d_in[2];
    const float* Wo = (const float*)d_in[3];
    const float* bo = (const float*)d_in[4];
    float* out = (float*)d_out;

    char* ws = (char*)d_ws;
    const size_t MB = 1024ull * 1024ull;
    _Float16* Qc = (_Float16*)(ws);              // 16 MB  [B,H,S,128]
    _Float16* Kc = (_Float16*)(ws + 16 * MB);    // 16 MB  [B,H,S,128]
    _Float16* Vt = (_Float16*)(ws + 32 * MB);    // 32 MB  [B,H,256,S]
    _Float16* Ot = (_Float16*)(ws + 64 * MB);    // 32 MB  [B,H,256,S]

    k1_qkvproj<<<NB * SEQ / 2, 256, 0, stream>>>(x, Wq, bq, Qc, Kc, Vt);
    k2_attn<<<NB * NH * (SEQ / 64), 128, 0, stream>>>(Qc, Kc, Vt, Ot);
    k3_outproj<<<NB * SEQ, 256, 0, stream>>>(Ot, Wo, bo, out);
}
